// ReservoirLayer_5506148073535
// MI455X (gfx1250) — matrix operand order repaired and checked by reading, not compiled, not run on hardware
//
#include <hip/hip_runtime.h>
#include <hip/hip_bf16.h>

// Echo-state network persistent kernel for MI455X (gfx1250, wave32).
// s_{t+1} = 0.05*s_t + 0.95*tanh([s_t | x_t] @ [Wrec^T ; Win^T] + bias)
//
// B=16 == WMMA M. bf16 WMMA (16x16x32) with f32 accumulate; f32 state.
// Weights are scan-invariant: each wave keeps its 18 B-fragments (144 VGPRs)
// register-resident for all 2048 steps. The 16 workgroups are dispatched as
// one CDNA5 workgroup-cluster when supported, so the per-step device-wide
// sync uses the hardware cluster barrier (s_barrier_signal/wait -3) instead
// of an atomic spin loop; an atomic grid barrier is the runtime fallback.
//
// Workspace layout (~192 KB):
//   [0,256)              : grid-barrier counter (memset to 0 each call)
//   [256, +64K)          : f32 state buffer 0   (16x1024)
//   [.. , +64K)          : f32 state buffer 1
//   [.. , +32K)          : bf16 state buffer 0  (16x1024)  (staging mirror)
//   [.. , +32K)          : bf16 state buffer 1

#define B_    16
#define T_    2048
#define F_    128
#define R_    1024
#define KTOT  1152          // R_ + F_
#define KPAD  1160          // LDS row stride in halves (stagger banks)
#define KHALF (KTOT / 2)    // 576: contiguous K range per wave of a pair
#define NWG   16
#define TPB   256
#define GAMMA_F 0.95f

#if defined(__has_attribute)
#  if __has_attribute(amdgpu_cluster_dims)
#    define CLUSTER_DIMS __attribute__((amdgpu_cluster_dims(NWG, 1, 1)))
#  endif
#endif
#ifndef CLUSTER_DIMS
#  define CLUSTER_DIMS
#endif

typedef __attribute__((ext_vector_type(16))) __bf16 v16bf;
typedef __attribute__((ext_vector_type(8)))  float  v8f;
typedef __attribute__((ext_vector_type(2)))  __bf16 v2bf;

union Frag16 { v16bf v; uint4 q[2]; };
union Pack4  { __bf16 h[4]; uint2 u; };

__device__ __forceinline__ float fast_tanh(float x) {
#if __has_builtin(__builtin_amdgcn_tanhf)
    return __builtin_amdgcn_tanhf(x);     // hw v_tanh_f32 on gfx1250
#else
    return tanhf(x);
#endif
}

__device__ __forceinline__ Pack4 cvt4(float4 v4) {
    Pack4 pk;
#if __has_builtin(__builtin_amdgcn_cvt_pk_bf16_f32)
    v2bf lo = __builtin_amdgcn_cvt_pk_bf16_f32(v4.x, v4.y);
    v2bf hh = __builtin_amdgcn_cvt_pk_bf16_f32(v4.z, v4.w);
    pk.u.x = __builtin_bit_cast(unsigned, lo);
    pk.u.y = __builtin_bit_cast(unsigned, hh);
#else
    pk.h[0] = (__bf16)v4.x; pk.h[1] = (__bf16)v4.y;
    pk.h[2] = (__bf16)v4.z; pk.h[3] = (__bf16)v4.w;
#endif
    return pk;
}

// In-cluster test: IB_STS2.Cluster_ID[3:0] (bits 9:6); 0 == not in a cluster.
// Uniform across the dispatch, so all WGs take the same barrier flavor.
__device__ __forceinline__ bool wave_in_cluster() {
#if __has_builtin(__builtin_amdgcn_s_cluster_barrier)
    return __builtin_amdgcn_s_getreg(28 | (6 << 6) | ((4 - 1) << 11)) != 0;
#else
    return false;
#endif
}

__device__ __forceinline__ void grid_barrier(unsigned* cnt, unsigned target, bool clus) {
#if __has_builtin(__builtin_amdgcn_s_cluster_barrier)
    if (clus) {
        // ISA-recommended: WG-sync first, then one wave per WG signals the
        // cluster user barrier (-3); trailing WG-sync releases other waves.
        __builtin_amdgcn_fence(__ATOMIC_RELEASE, "agent");   // flush my stores
        __syncthreads();
        if ((threadIdx.x >> 5) == 0)
            __builtin_amdgcn_s_cluster_barrier();            // signal -3; wait -3
        __syncthreads();
        __builtin_amdgcn_fence(__ATOMIC_ACQUIRE, "agent");   // invalidate WGP$
        return;
    }
#endif
    __syncthreads();
    if (threadIdx.x == 0) {
        __builtin_amdgcn_fence(__ATOMIC_RELEASE, "agent");
        __hip_atomic_fetch_add(cnt, 1u, __ATOMIC_RELAXED, __HIP_MEMORY_SCOPE_AGENT);
        while (__hip_atomic_load(cnt, __ATOMIC_RELAXED, __HIP_MEMORY_SCOPE_AGENT) < target)
            __builtin_amdgcn_s_sleep(1);
    }
    __syncthreads();
    __builtin_amdgcn_fence(__ATOMIC_ACQUIRE, "agent");
}

__global__ __launch_bounds__(TPB, 1) CLUSTER_DIMS
void esn_persistent_kernel(const float* __restrict__ inp,    // (B,T,F)
                           const float* __restrict__ win,    // (R,F)
                           const float* __restrict__ wrec,   // (R,R)
                           const float* __restrict__ bias,   // (R,1)
                           const float* __restrict__ s0,     // (R,1)
                           float* __restrict__ out,          // (B,T,R)
                           unsigned char* __restrict__ ws) {
    unsigned* cnt = (unsigned*)ws;
    float*  S32[2];
    S32[0] = (float*)(ws + 256);
    S32[1] = (float*)(ws + 256 + (size_t)B_ * R_ * 4);
    __bf16* Sbf[2];
    Sbf[0] = (__bf16*)(ws + 256 + 2 * (size_t)B_ * R_ * 4);
    Sbf[1] = (__bf16*)(ws + 256 + 2 * (size_t)B_ * R_ * 4 + (size_t)B_ * R_ * 2);

    __shared__ __bf16 ldsA[B_ * KPAD];   // staged [s | x_t] in bf16
    __shared__ float  red[4 * 256];      // K-split partial reduction, 4 tiles

    const int tid  = threadIdx.x;
    const int lane = tid & 31;
    const int w    = tid >> 5;           // wave 0..7
    const int hi   = lane >> 4;          // lane half: 0 or 1
    const int n    = lane & 15;          // WMMA column (and A-row) index
    const int tile = w >> 1;             // N-tile 0..3 within this WG
    const int p    = w & 1;              // K-split half (contiguous)
    const int col  = blockIdx.x * 64 + tile * 16 + n;   // global output column
    const bool clus = wave_in_cluster();

    // ---- One-time: convert this lane's B-fragments to bf16 registers ----
    // B element (k, col): k < R_ -> wrec[col][k], else win[col][k - R_].
    // Per-lane fragment data is a contiguous 16-wide K run (never straddles
    // the R_ boundary since tiles are 32-aligned).
    Frag16 bw[18];
    #pragma unroll
    for (int kt = 0; kt < 18; ++kt) {
        const int k0 = p * KHALF + kt * 32 + hi * 16;
        const float* src = (k0 < R_) ? (wrec + (size_t)col * R_ + k0)
                                     : (win + (size_t)col * F_ + (k0 - R_));
        Pack4 p0 = cvt4(*(const float4*)(src));
        Pack4 p1 = cvt4(*(const float4*)(src + 4));
        Pack4 p2 = cvt4(*(const float4*)(src + 8));
        Pack4 p3 = cvt4(*(const float4*)(src + 12));
        bw[kt].q[0] = uint4{p0.u.x, p0.u.y, p1.u.x, p1.u.y};
        bw[kt].q[1] = uint4{p2.u.x, p2.u.y, p3.u.x, p3.u.y};
    }

    // ---- Pre-pass: initial state (f32 + bf16 mirror), then grid sync ----
    {
        const int gs   = NWG * TPB;
        const int gtid = blockIdx.x * TPB + tid;
        for (int i = gtid; i < B_ * R_; i += gs) {
            float v = s0[i & (R_ - 1)];      // broadcast start state over batch
            S32[0][i] = v;
            Sbf[0][i] = (__bf16)v;
        }
    }
    unsigned target = NWG;
    grid_barrier(cnt, target, clus);

    const float bias_c = bias[col];
    const __bf16* abase = ldsA + n * KPAD + hi * 16 + p * KHALF;

    for (int t = 0; t < T_; ++t) {
        const float*  Scur  = S32[t & 1];
        float*        Snxt  = S32[(t + 1) & 1];
        const __bf16* Scbf  = Sbf[t & 1];
        __bf16*       Snbf  = Sbf[(t + 1) & 1];

        // ---- Stage A = [s | x_t] into LDS ----
        // State part: pure bf16 b128 copy (no conversion on the hot path).
        for (int i8 = tid; i8 < (B_ * R_) / 8; i8 += TPB) {
            const int m = i8 >> 7;             // 128 uint4 per 1024-half row
            const int k = (i8 & 127) * 8;
            *(uint4*)(&ldsA[m * KPAD + k]) = *(const uint4*)(Scbf + m * R_ + k);
        }
        // Input slice: f32 -> bf16 (only 16x128 elements).
        const float* xrow = inp + (size_t)t * F_;
        for (int i4 = tid; i4 < (B_ * F_) / 4; i4 += TPB) {
            const int m = i4 >> 5;             // 32 float4 per 128-f32 row
            const int k = (i4 & 31) * 4;
            float4 v4 = *(const float4*)(xrow + (size_t)m * T_ * F_ + k);
            *(uint2*)(&ldsA[m * KPAD + R_ + k]) = cvt4(v4).u;
        }
        __syncthreads();

        // ---- GEMM: A from LDS (SW-pipelined), B register-resident.
        // Dual accumulators break the C-dependence chain between WMMAs.
        v8f acc0 = {}, acc1 = {};
        Frag16 af[4];
        #pragma unroll
        for (int i = 0; i < 4; ++i) {
            af[i].q[0] = *(const uint4*)(abase + i * 32);
            af[i].q[1] = *(const uint4*)(abase + i * 32 + 8);
        }
        #pragma unroll
        for (int kt = 0; kt < 18; ++kt) {
            const int slot = kt & 3;
            if (kt & 1)
                acc1 = __builtin_amdgcn_wmma_f32_16x16x32_bf16(
                    false, af[slot].v, false, bw[kt].v, (short)0, acc1, false, false);
            else
                acc0 = __builtin_amdgcn_wmma_f32_16x16x32_bf16(
                    false, af[slot].v, false, bw[kt].v, (short)0, acc0, false, false);
            const int nxt = kt + 4;
            if (nxt < 18) {   // refill the slot; load-pairs stay in flight
                af[slot].q[0] = *(const uint4*)(abase + nxt * 32);
                af[slot].q[1] = *(const uint4*)(abase + nxt * 32 + 8);
            }
        }

        // ---- Reduce wave-pair partials through LDS, then epilogue ----
        if (p == 1) {
            #pragma unroll
            for (int vv = 0; vv < 8; ++vv)
                red[tile * 256 + (8 * hi + vv) * 16 + n] = acc0[vv] + acc1[vv];
        }
        __syncthreads();
        if (p == 0) {
            #pragma unroll
            for (int vv = 0; vv < 8; ++vv) {
                const int m = 8 * hi + vv;                    // batch row
                float pre  = acc0[vv] + acc1[vv]
                           + red[tile * 256 + m * 16 + n] + bias_c;
                float th   = fast_tanh(pre);
                float sold = Scur[m * R_ + col];
                float sn   = (1.0f - GAMMA_F) * sold + GAMMA_F * th;
                Snxt[m * R_ + col] = sn;
                Snbf[m * R_ + col] = (__bf16)sn;
                out[(size_t)m * T_ * R_ + (size_t)t * R_ + col] = sn;
            }
        }

        target += NWG;
        grid_barrier(cnt, target, clus);   // all WGs see Snxt/Snbf next step
    }
}

extern "C" void kernel_launch(void* const* d_in, const int* in_sizes, int n_in,
                              void* d_out, int out_size, void* d_ws, size_t ws_size,
                              hipStream_t stream) {
    const float* inputs  = (const float*)d_in[0];   // (16,2048,128)
    const float* win     = (const float*)d_in[1];   // (1024,128)
    const float* wrec    = (const float*)d_in[2];   // (1024,1024)
    const float* bias    = (const float*)d_in[3];   // (1024,1)
    const float* s0      = (const float*)d_in[4];   // (1024,1)
    float* out = (float*)d_out;

    // Zero the grid-barrier counter region (graph-capture-safe, deterministic).
    hipMemsetAsync(d_ws, 0, 256, stream);

    esn_persistent_kernel<<<dim3(NWG), dim3(TPB), 0, stream>>>(
        inputs, win, wrec, bias, s0, out, (unsigned char*)d_ws);
}
